// ModelBaseline_49357764165987
// MI455X (gfx1250) — compile-verified, matrix operand order checked by reading, not run on hardware
//
#include <hip/hip_runtime.h>
#include <hip/hip_bf16.h>

typedef __attribute__((ext_vector_type(16))) _Float16 v16h;
typedef __attribute__((ext_vector_type(8)))  _Float16 v8h;
typedef __attribute__((ext_vector_type(8)))  float    v8f;
typedef __attribute__((ext_vector_type(4)))  int      v4i;

// pointer types matching the async-load builtin's expected signature:
// (int4 __device__*, int4 __shared__*, int, int)
typedef __attribute__((address_space(1))) v4i* gas_v4i_ptr;
typedef __attribute__((address_space(3))) v4i* las_v4i_ptr;

#define B_ROWS   16384
#define L_SEQ    2048
#define N_COD    64
#define N_TIS    30
#define D_IN     128   // 64 emb + 64 freq
#define D_HID    256
#define MLP_WAVES 8    // waves per k_mlp block; each wave owns one 16-row tile

// ---------------------------------------------------------------------------
// Kernel 0a: norm-clip the 30x64 embedding table, store as f16.
// ---------------------------------------------------------------------------
__global__ void k_emb_scale(const float* __restrict__ emb_table,
                            _Float16* __restrict__ embS)
{
    __shared__ float sq[64];
    const int r = blockIdx.x;     // tissue row
    const int t = threadIdx.x;    // 0..63
    const float v = emb_table[r * 64 + t];
    sq[t] = v * v;
    __syncthreads();
    float s = 0.0f;
#pragma unroll
    for (int i = 0; i < 64; ++i) s += sq[i];      // LDS broadcast reads
    const float norm  = sqrtf(s);
    const float scale = (norm > 1.0f) ? (1.0f / (norm + 1e-7f)) : 1.0f;
    embS[r * 64 + t] = (_Float16)(v * scale);
}

// ---------------------------------------------------------------------------
// Kernel 0b: convert W1 (256x128 f32, row-major over k) to f16.
// ---------------------------------------------------------------------------
__global__ void k_w1_cvt(const float* __restrict__ W1,
                         _Float16* __restrict__ w1h)
{
    const int i = blockIdx.x * blockDim.x + threadIdx.x;
    if (i < D_HID * D_IN) w1h[i] = (_Float16)W1[i];
}

// ---------------------------------------------------------------------------
// Kernel 1 (the bandwidth-bound pass): per-row codon histogram -> freq,
// fused with embedding copy. One block (8 wave32) per batch row.
// 128 MB of rna_data streamed once with NT b128 loads (~5.5 us at 23.3 TB/s).
// Per-wave LDS histogram replicas (72-int stride staggers banks); total valid
// codons recovered as 2048 - count(bin 0), so no sum-reduction needed.
// ---------------------------------------------------------------------------
__global__ __launch_bounds__(256) void k_hist(const int* __restrict__ rna,
                                              const int* __restrict__ tissue,
                                              const _Float16* __restrict__ embS,
                                              _Float16* __restrict__ xh)
{
    __shared__ unsigned int hist[8][72];    // 8 per-wave replicas, padded
    __shared__ unsigned int comb[72];

    const int row = blockIdx.x;
    const int t   = threadIdx.x;            // 0..255
    const int w   = t >> 5;                 // wave id 0..7 (wave32)

    for (int i = t; i < 8 * 72; i += 256) (&hist[0][0])[i] = 0u;
    __syncthreads();

    const v4i* __restrict__ p4 = (const v4i*)(rna + (size_t)row * L_SEQ);
#pragma unroll
    for (int i = 0; i < 2; ++i) {
        const v4i q = __builtin_nontemporal_load(p4 + t + i * 256);
        atomicAdd(&hist[w][q.x], 1u);
        atomicAdd(&hist[w][q.y], 1u);
        atomicAdd(&hist[w][q.z], 1u);
        atomicAdd(&hist[w][q.w], 1u);
    }
    __syncthreads();

    if (t < 65) {
        unsigned int c = 0;
#pragma unroll
        for (int ww = 0; ww < 8; ++ww) c += hist[ww][t];
        comb[t] = c;
    }
    __syncthreads();

    if (t < 64) {
        const float inv = 1.0f / (float)(L_SEQ - (int)comb[0]);
        xh[(size_t)row * D_IN + 64 + t] = (_Float16)((float)comb[1 + t] * inv);
    } else if (t < 128) {
        const int j   = t - 64;
        const int tid = tissue[row];
        xh[(size_t)row * D_IN + j] = embS[tid * 64 + j];
    }
}

// ---------------------------------------------------------------------------
// Kernel 2: fused MLP.  8 waves per block; each wave owns a 16-row M-tile.
// W1 (64 KB f16), b1, W2 are staged in LDS once per block (async global->LDS
// when the toolchain exposes it), so the inner loop is ds_load_b128 + wmma
// instead of L2-latency global loads in front of every WMMA.
//   h   = relu(x @ W1^T + b1)   : v_wmma_f32_16x16x32_f16, K=128 = 4 x K32
//   out = sigmoid(h @ W2^T + b2): folded into the N-tile loop, h never hits
//                                 memory; half-wave shfl_xor reduction.
// ---------------------------------------------------------------------------
__global__ __launch_bounds__(256) void k_mlp(const _Float16* __restrict__ xh,
                                             const _Float16* __restrict__ w1h,
                                             const float* __restrict__ b1,
                                             const float* __restrict__ W2,
                                             const float* __restrict__ b2,
                                             float* __restrict__ out)
{
    __shared__ _Float16 w1s[D_HID * D_IN];   // 64 KB  (320 KB WGP LDS: fits)
    __shared__ float    b1s[D_HID];
    __shared__ float    w2s[D_HID];

    const int t    = threadIdx.x;            // 0..255
    const int wave = t >> 5;
    const int lane = t & 31;
    const int half = lane >> 4;              // K-chunk select (wave32 halves)
    const int lm   = lane & 15;

    // ---- stage W1h (4096 x 16B), b1, W2 into LDS ----
#if defined(__HIP_DEVICE_COMPILE__) && __has_builtin(__builtin_amdgcn_global_load_async_to_lds_b128)
    for (int i = t; i < (D_HID * D_IN) / 8; i += 256) {
        __builtin_amdgcn_global_load_async_to_lds_b128(
            (gas_v4i_ptr)(w1h + (size_t)i * 8),
            (las_v4i_ptr)(w1s + (size_t)i * 8),
            /*offset=*/0, /*cpol=*/0);
    }
    for (int i = t; i < D_HID; i += 256) { b1s[i] = b1[i]; w2s[i] = W2[i]; }
#if __has_builtin(__builtin_amdgcn_s_wait_asynccnt)
    __builtin_amdgcn_s_wait_asynccnt(0);
#else
    asm volatile("s_wait_asynccnt 0x0" ::: "memory");
#endif
    __syncthreads();
#else
    for (int i = t; i < (D_HID * D_IN) / 8; i += 256)
        *(v8h*)(w1s + (size_t)i * 8) = *(const v8h*)(w1h + (size_t)i * 8);
    for (int i = t; i < D_HID; i += 256) { b1s[i] = b1[i]; w2s[i] = W2[i]; }
    __syncthreads();
#endif

    const int m0 = blockIdx.x * (MLP_WAVES * 16) + wave * 16;

    // ---- load the 4 A fragments (rows m0..m0+15, K=0..127) from global ----
    // ISA 7.12.2 wave32 A layout: lane holds row M=(l&15); elems 0..7 hold
    // K = kb*32 + (l>>4)*8 + e, elems 8..15 the same +16 -> two 16B loads.
    v16h A[4];
    {
        const size_t arow = (size_t)(m0 + lm) * D_IN;
#pragma unroll
        for (int kb = 0; kb < 4; ++kb) {
            const size_t base = arow + kb * 32 + half * 8;
            const v8h lo = *(const v8h*)(xh + base);
            const v8h hi = *(const v8h*)(xh + base + 16);
#pragma unroll
            for (int e = 0; e < 8; ++e) { A[kb][e] = lo[e]; A[kb][e + 8] = hi[e]; }
        }
    }

    float acc2[8];
#pragma unroll
    for (int r = 0; r < 8; ++r) acc2[r] = 0.0f;

    // ---- 16 N-tiles of the 256 hidden units; B fragments from LDS ----
    for (int nt = 0; nt < 16; ++nt) {
        const int n = nt * 16 + lm;          // hidden unit owned by this lane
        v8f c = {};
#pragma unroll
        for (int kb = 0; kb < 4; ++kb) {
            // B fragment: B[k][n] = W1[n][k]; same K mapping as A per lane.
            const size_t base = (size_t)n * D_IN + kb * 32 + half * 8;
            const v8h lo = *(const v8h*)(w1s + base);   // ds_load_b128
            const v8h hi = *(const v8h*)(w1s + base + 16);
            v16h Bf;
#pragma unroll
            for (int e = 0; e < 8; ++e) { Bf[e] = lo[e]; Bf[e + 8] = hi[e]; }
            c = __builtin_amdgcn_wmma_f32_16x16x32_f16(
                    /*neg_a=*/false, A[kb], /*neg_b=*/false, Bf,
                    /*c_mod=*/(short)0, c, /*reuse_a=*/false, /*reuse_b=*/false);
        }
        const float bias = b1s[n];
        const float w2n  = w2s[n];
#pragma unroll
        for (int r = 0; r < 8; ++r) {
            float h = c[r] + bias;           // h[m0 + r + half*8][n]
            h = (h > 0.0f) ? h : 0.0f;       // relu
            acc2[r] += h * w2n;              // layer-2 partial dot
        }
    }

    // ---- reduce layer-2 partials across each 16-lane half ----
#pragma unroll
    for (int off = 8; off >= 1; off >>= 1) {
#pragma unroll
        for (int r = 0; r < 8; ++r)
            acc2[r] += __shfl_xor(acc2[r], off);   // stays within lane half
    }

    if (lm == 0) {
        const float bb = b2[0];
#pragma unroll
        for (int r = 0; r < 8; ++r) {
            const float z = acc2[r] + bb;
            out[m0 + half * 8 + r] = 1.0f / (1.0f + expf(-z));
        }
    }
}

// ---------------------------------------------------------------------------
// Launch
// ---------------------------------------------------------------------------
extern "C" void kernel_launch(void* const* d_in, const int* in_sizes, int n_in,
                              void* d_out, int out_size, void* d_ws, size_t ws_size,
                              hipStream_t stream)
{
    const int*   rna    = (const int*)  d_in[0];   // (16384, 2048) int32
    const int*   tissue = (const int*)  d_in[1];   // (16384,) int32
    const float* embT   = (const float*)d_in[2];   // (30, 64) f32
    const float* W1     = (const float*)d_in[3];   // (256, 128) f32
    const float* b1     = (const float*)d_in[4];   // (256,) f32
    const float* W2     = (const float*)d_in[5];   // (1, 256) f32
    const float* b2     = (const float*)d_in[6];   // (1,) f32
    float*       out    = (float*)d_out;           // (16384, 1) f32

    // workspace layout (all 256B-aligned)
    char* ws = (char*)d_ws;
    _Float16* xh   = (_Float16*)ws;                              // 16384*128*2 = 4 MiB
    _Float16* w1h  = (_Float16*)(ws + (size_t)4 * 1024 * 1024);  // 256*128*2 = 64 KiB
    _Float16* embS = (_Float16*)(ws + (size_t)4 * 1024 * 1024 + 65536); // 30*64*2

    k_emb_scale<<<N_TIS, 64, 0, stream>>>(embT, embS);
    k_w1_cvt<<<(D_HID * D_IN + 255) / 256, 256, 0, stream>>>(W1, w1h);
    k_hist<<<B_ROWS, 256, 0, stream>>>(rna, tissue, embS, xh);
    k_mlp<<<B_ROWS / (16 * MLP_WAVES), 256, 0, stream>>>(xh, w1h, b1, W2, b2, out);
}
